// CategoryBranch_5729486372935
// MI455X (gfx1250) — matrix operand order repaired and checked by reading, not compiled, not run on hardware
//
#include <hip/hip_runtime.h>
#include <hip/hip_bf16.h>
#include <math.h>

// ---------------------------------------------------------------------------
// CDNA5 / gfx1250: conv + 3-layer GRU (B=64, H=512, T=256) with bf16 WMMA.
//   1) conv -> xs[T,B,3968] bf16
//   2) gi0 = xs @ w_ih0^T + b_ih0 : one big WMMA GEMM (61% of FLOPs, parallel)
//   3) ONE persistent kernel runs the whole 256-step recurrence:
//        - 32 WGs x 128 thr; WG j owns gate columns {j16, 512+j16, 1024+j16}
//        - all recurrent weight slices cached in LDS (240 KB of the 320 KB WGP LDS)
//        - hidden state kept in fp32 + bf16 mirror (b128 A-tile loads, no cvt)
//        - device-scope atomic grid barrier between layers (3 per step)
// ---------------------------------------------------------------------------

typedef __bf16 bf16_t;
typedef __attribute__((ext_vector_type(16))) __bf16 v16bf;
typedef __attribute__((ext_vector_type(8)))  float  v8f;

#define B_    64
#define T_    256
#define H_    512
#define G_    1536      // 3*H
#define RIN_  3968      // 64 filters * 62 rows
#define BH_   (B_ * H_)
#define NWG_  32
#define LDS_SLOT_HALFS (16 * 512)                 // one 16-col x 512-K slice
#define LDS_BYTES (15 * LDS_SLOT_HALFS * 2)       // 245760 B

__device__ __forceinline__ v8f wmma_bf16(v16bf a, v16bf b, v8f c) {
  return __builtin_amdgcn_wmma_f32_16x16x32_bf16(false, a, false, b, (short)0, c,
                                                 false, false);
}

// ---- A tile (16x32 bf16) from bf16 matrix [rows, ld], ld multiple of 8.
// A layout (matches B): lanes 0-15: M=lane, K=0..15;
//               lanes 16-31: M=lane-16, K=16..31.
__device__ __forceinline__ v16bf load_a_bf16(const bf16_t* __restrict__ A,
                                             int mbase, int lane, int kofs, int ld) {
  int m  = mbase + (lane & 15);
  int kb = (lane >> 4) * 16;
  const bf16_t* p = A + (size_t)m * ld + kofs + kb;
  union { v16bf v; uint4 u[2]; } t;
  t.u[0] = *reinterpret_cast<const uint4*>(p);
  t.u[1] = *reinterpret_cast<const uint4*>(p + 8);
  return t.v;
}

// ---- B tile (32x16) from a weight slice stored [16 cols x 512 K] (LDS or global).
// Layout: lane n = lane&15, lanes 0-15 K=0..15 (vi = K 2i,2i+1), lanes 16-31 K=16..31.
__device__ __forceinline__ v16bf load_b_slice(const bf16_t* base, int lane, int kofs) {
  int c  = lane & 15;
  int kb = (lane >> 4) * 16;
  const bf16_t* p = base + c * H_ + kofs + kb;
  union { v16bf v; uint4 u[2]; } t;
  t.u[0] = *reinterpret_cast<const uint4*>(p);
  t.u[1] = *reinterpret_cast<const uint4*>(p + 8);
  return t.v;
}

// ---- B tile (32x16) from full weight W[G_, ld] row-major (big GEMM).
__device__ __forceinline__ v16bf load_b_bf16(const bf16_t* __restrict__ W,
                                             int nbase, int lane, int kofs, int ld) {
  int n  = nbase + (lane & 15);
  int kb = (lane >> 4) * 16;
  const bf16_t* p = W + (size_t)n * ld + kofs + kb;
  union { v16bf v; uint4 u[2]; } t;
  t.u[0] = *reinterpret_cast<const uint4*>(p);
  t.u[1] = *reinterpret_cast<const uint4*>(p + 8);
  return t.v;
}

__device__ __forceinline__ float sigmoidf_(float x) { return 1.0f / (1.0f + __expf(-x)); }

// ---------------------------------------------------------------------------
// Kernel 1: conv(1->64, 3x3, pad (0,1)) + bias -> xs[t][b][k=f*62+r] bf16.
// ---------------------------------------------------------------------------
__global__ void conv_to_xs(const float* __restrict__ x, const float* __restrict__ cw,
                           const float* __restrict__ cb, bf16_t* __restrict__ xs) {
  int idx = blockIdx.x * 256 + threadIdx.x;
  const int total = T_ * B_ * RIN_;
  if (idx >= total) return;
  int k  = idx % RIN_;
  int bt = idx / RIN_;
  int b  = bt % B_;
  int t  = bt / B_;
  int f  = k / 62;
  int r  = k % 62;
  const float* xp = x + ((size_t)b * 64 + r) * 256 + t;
  const float* w  = cw + f * 9;
  float s = cb[f];
#pragma unroll
  for (int kh = 0; kh < 3; ++kh) {
    const float* row = xp + kh * 256;
    if (t > 0)   s += row[-1] * w[kh * 3 + 0];
                 s += row[ 0] * w[kh * 3 + 1];
    if (t < 255) s += row[ 1] * w[kh * 3 + 2];
  }
  xs[idx] = (__bf16)s;
}

__global__ void f32_to_bf16_k(const float* __restrict__ in, bf16_t* __restrict__ out, int n) {
  int i = blockIdx.x * 256 + threadIdx.x;
  if (i < n) out[i] = (__bf16)in[i];
}

// ---------------------------------------------------------------------------
// Kernel 3: gi0[16384,1536] = xs[16384,3968] @ w_ih0^T + b_ih0
// 8 waves/WG (2Mx4N), 32x32 per wave.  Grid (256, 12).
// ---------------------------------------------------------------------------
__global__ __launch_bounds__(256)
void gemm_gi0(const bf16_t* __restrict__ X, const bf16_t* __restrict__ W,
              const float* __restrict__ bias, float* __restrict__ C) {
  int wave = threadIdx.x >> 5, lane = threadIdx.x & 31;
  int wm = wave >> 2, wn = wave & 3;
  int mbase = blockIdx.x * 64 + wm * 32;
  int nbase = blockIdx.y * 128 + wn * 32;
  v8f acc00 = {}, acc01 = {}, acc10 = {}, acc11 = {};
#pragma unroll 2
  for (int k = 0; k < RIN_; k += 32) {
    v16bf a0 = load_a_bf16(X, mbase,      lane, k, RIN_);
    v16bf a1 = load_a_bf16(X, mbase + 16, lane, k, RIN_);
    v16bf b0 = load_b_bf16(W, nbase,      lane, k, RIN_);
    v16bf b1 = load_b_bf16(W, nbase + 16, lane, k, RIN_);
    acc00 = wmma_bf16(a0, b0, acc00);
    acc01 = wmma_bf16(a0, b1, acc01);
    acc10 = wmma_bf16(a1, b0, acc10);
    acc11 = wmma_bf16(a1, b1, acc11);
  }
  int rofs = (lane >> 4) * 8, col = lane & 15;
  v8f accs[2][2] = {{acc00, acc01}, {acc10, acc11}};
#pragma unroll
  for (int mi = 0; mi < 2; ++mi)
#pragma unroll
    for (int ni = 0; ni < 2; ++ni) {
      int nc = nbase + ni * 16 + col;
      float bia = bias[nc];
      float* cp = C + (size_t)(mbase + mi * 16 + rofs) * G_ + nc;
#pragma unroll
      for (int v = 0; v < 8; ++v) cp[(size_t)v * G_] = accs[mi][ni][v] + bia;
    }
}

// ---------------------------------------------------------------------------
// Persistent GRU recurrence.
// ---------------------------------------------------------------------------
__device__ __forceinline__ void grid_barrier(unsigned* cnt, unsigned* gen) {
  __syncthreads();
  if (threadIdx.x == 0) {
    __threadfence();
    unsigned g = __hip_atomic_load(gen, __ATOMIC_ACQUIRE, __HIP_MEMORY_SCOPE_AGENT);
    unsigned my = __hip_atomic_fetch_add(cnt, 1u, __ATOMIC_ACQ_REL, __HIP_MEMORY_SCOPE_AGENT);
    if (my == NWG_ - 1u) {
      __hip_atomic_store(cnt, 0u, __ATOMIC_RELAXED, __HIP_MEMORY_SCOPE_AGENT);
      __hip_atomic_fetch_add(gen, 1u, __ATOMIC_ACQ_REL, __HIP_MEMORY_SCOPE_AGENT);
    } else {
      while (__hip_atomic_load(gen, __ATOMIC_ACQUIRE, __HIP_MEMORY_SCOPE_AGENT) == g)
        __builtin_amdgcn_s_sleep(1);
    }
  }
  __syncthreads();
}

// One layer's gate GEMM; weight slices live in LDS (matIH/matHH = slot index 0..4).
template <bool HAS_IH>
__device__ __forceinline__ void layer_gemm(const bf16_t* lds, int matIH, int matHH,
                                           const bf16_t* __restrict__ xbf,
                                           const bf16_t* __restrict__ hbf,
                                           int mbase, int lane,
                                           v8f& accr, v8f& accz, v8f& accin, v8f& acchn) {
  const bf16_t* hR = lds + (matHH * 3 + 0) * LDS_SLOT_HALFS;
  const bf16_t* hZ = lds + (matHH * 3 + 1) * LDS_SLOT_HALFS;
  const bf16_t* hN = lds + (matHH * 3 + 2) * LDS_SLOT_HALFS;
  const bf16_t* iR = lds + (matIH * 3 + 0) * LDS_SLOT_HALFS;
  const bf16_t* iZ = lds + (matIH * 3 + 1) * LDS_SLOT_HALFS;
  const bf16_t* iN = lds + (matIH * 3 + 2) * LDS_SLOT_HALFS;
#pragma unroll 2
  for (int k = 0; k < H_; k += 32) {
    v16bf ah = load_a_bf16(hbf, mbase, lane, k, H_);
    accr  = wmma_bf16(ah, load_b_slice(hR, lane, k), accr);
    accz  = wmma_bf16(ah, load_b_slice(hZ, lane, k), accz);
    acchn = wmma_bf16(ah, load_b_slice(hN, lane, k), acchn);
    if (HAS_IH) {
      v16bf ax = load_a_bf16(xbf, mbase, lane, k, H_);
      accr  = wmma_bf16(ax, load_b_slice(iR, lane, k), accr);
      accz  = wmma_bf16(ax, load_b_slice(iZ, lane, k), accz);
      accin = wmma_bf16(ax, load_b_slice(iN, lane, k), accin);
    }
  }
}

__device__ __forceinline__ void gru_elem(const v8f& accr, const v8f& accz,
                                         const v8f& accin, const v8f& acchn,
                                         bool useGi, const float* __restrict__ gi_t,
                                         const float* __restrict__ bih,
                                         const float* __restrict__ bhh,
                                         const float* __restrict__ hin_f,
                                         float* __restrict__ hout_f,
                                         bf16_t* __restrict__ hout_bf,
                                         float* __restrict__ outp,
                                         int mbase, int lane, int j) {
  int rofs = (lane >> 4) * 8, col = lane & 15;
  int nr = j * 16 + col, nz = H_ + nr, nn = 2 * H_ + nr;
  int hcol = j * 16 + col;
  float bhr = bhh[nr], bhz = bhh[nz], bhn = bhh[nn];
  float bir = 0.f, biz = 0.f, bin = 0.f;
  if (!useGi) { bir = bih[nr]; biz = bih[nz]; bin = bih[nn]; }
#pragma unroll
  for (int v = 0; v < 8; ++v) {
    int row = mbase + rofs + v;
    float gr = accr[v] + bhr, gz = accz[v] + bhz, ghn = acchn[v] + bhn, gin;
    if (useGi) {
      const float* g = gi_t + (size_t)row * G_;
      gr += g[nr]; gz += g[nz]; gin = g[nn];
    } else {
      gr += bir; gz += biz; gin = accin[v] + bin;
    }
    float r = sigmoidf_(gr);
    float z = sigmoidf_(gz);
    float nv = tanhf(gin + r * ghn);
    float hold = hin_f[(size_t)row * H_ + hcol];
    float hnew = (1.0f - z) * nv + z * hold;
    hout_f[(size_t)row * H_ + hcol]  = hnew;
    hout_bf[(size_t)row * H_ + hcol] = (__bf16)hnew;
    if (outp) outp[(size_t)row * H_ + hcol] = hnew;
  }
}

__global__ __launch_bounds__(128)
void gru_persistent(const bf16_t* __restrict__ whh0, const bf16_t* __restrict__ wih1,
                    const bf16_t* __restrict__ whh1, const bf16_t* __restrict__ wih2,
                    const bf16_t* __restrict__ whh2,
                    const float* __restrict__ bhh0,
                    const float* __restrict__ bih1, const float* __restrict__ bhh1,
                    const float* __restrict__ bih2, const float* __restrict__ bhh2,
                    const float* __restrict__ gi0,
                    float* __restrict__ stf, bf16_t* __restrict__ stbf,
                    float* __restrict__ out, unsigned* __restrict__ bar) {
  extern __shared__ bf16_t lds[];
  int j = blockIdx.x;
  int wave = threadIdx.x >> 5, lane = threadIdx.x & 31;
  int mbase = wave * 16;

  // --- cache this WG's weight slices in LDS (15 slices x 16 cols x 512 K) ---
  // slot s = mat*3 + gate; mats: 0=whh0 1=wih1 2=whh1 3=wih2 4=whh2
  const bf16_t* msrc[5] = {whh0, wih1, whh1, wih2, whh2};
  for (int u = threadIdx.x; u < 15 * 16 * 64; u += 128) {   // uint4 = 8 halves
    int k8   = u & 63;
    int rowi = u >> 6;         // 0..239
    int s    = rowi >> 4;      // 0..14
    int c    = rowi & 15;
    int g    = s % 3, m = s / 3;
    int grow = g * H_ + j * 16 + c;
    reinterpret_cast<uint4*>(lds)[u] =
        reinterpret_cast<const uint4*>(msrc[m] + (size_t)grow * H_)[k8];
  }
  __syncthreads();

  unsigned* cnt = bar;
  unsigned* gen = bar + 32;    // separate cacheline

  for (int t = 0; t < T_; ++t) {
    int pi = t & 1, po = pi ^ 1;
    const float* gi_t = gi0 + (size_t)t * B_ * G_;
    // ---- layer 0 (ih part precomputed in gi0) ----
    {
      v8f accr = {}, accz = {}, accin = {}, acchn = {};
      layer_gemm<false>(lds, 0, 0, nullptr, stbf + (size_t)(0 * 2 + pi) * BH_,
                        mbase, lane, accr, accz, accin, acchn);
      gru_elem(accr, accz, accin, acchn, true, gi_t, nullptr, bhh0,
               stf + (size_t)(0 * 2 + pi) * BH_, stf + (size_t)(0 * 2 + po) * BH_,
               stbf + (size_t)(0 * 2 + po) * BH_, nullptr, mbase, lane, j);
    }
    grid_barrier(cnt, gen);
    // ---- layer 1: x = new c1 ----
    {
      v8f accr = {}, accz = {}, accin = {}, acchn = {};
      layer_gemm<true>(lds, 1, 2, stbf + (size_t)(0 * 2 + po) * BH_,
                       stbf + (size_t)(1 * 2 + pi) * BH_,
                       mbase, lane, accr, accz, accin, acchn);
      gru_elem(accr, accz, accin, acchn, false, nullptr, bih1, bhh1,
               stf + (size_t)(1 * 2 + pi) * BH_, stf + (size_t)(1 * 2 + po) * BH_,
               stbf + (size_t)(1 * 2 + po) * BH_, nullptr, mbase, lane, j);
    }
    grid_barrier(cnt, gen);
    // ---- layer 2: x = new c2, also writes the output slice ----
    {
      v8f accr = {}, accz = {}, accin = {}, acchn = {};
      layer_gemm<true>(lds, 3, 4, stbf + (size_t)(1 * 2 + po) * BH_,
                       stbf + (size_t)(2 * 2 + pi) * BH_,
                       mbase, lane, accr, accz, accin, acchn);
      gru_elem(accr, accz, accin, acchn, false, nullptr, bih2, bhh2,
               stf + (size_t)(2 * 2 + pi) * BH_, stf + (size_t)(2 * 2 + po) * BH_,
               stbf + (size_t)(2 * 2 + po) * BH_, out + (size_t)t * BH_, mbase, lane, j);
    }
    grid_barrier(cnt, gen);
  }
}

// ---------------------------------------------------------------------------
extern "C" void kernel_launch(void* const* d_in, const int* in_sizes, int n_in,
                              void* d_out, int out_size, void* d_ws, size_t ws_size,
                              hipStream_t stream) {
  const float* x      = (const float*)d_in[0];
  const float* hinit[3] = {(const float*)d_in[1], (const float*)d_in[2], (const float*)d_in[3]};
  const float* conv_w = (const float*)d_in[4];
  const float* conv_b = (const float*)d_in[5];
  const float* w_ih[3] = {(const float*)d_in[6],  (const float*)d_in[10], (const float*)d_in[14]};
  const float* w_hh[3] = {(const float*)d_in[7],  (const float*)d_in[11], (const float*)d_in[15]};
  const float* b_ih[3] = {(const float*)d_in[8],  (const float*)d_in[12], (const float*)d_in[16]};
  const float* b_hh[3] = {(const float*)d_in[9],  (const float*)d_in[13], (const float*)d_in[17]};
  float* out = (float*)d_out;

  char* base = (char*)d_ws;
  size_t off = 0;
  auto carve = [&](size_t bytes) -> char* {
    char* p = base + off;
    off = (off + bytes + 255) & ~(size_t)255;
    return p;
  };
  bf16_t* xs_bf   = (bf16_t*)carve((size_t)T_ * B_ * RIN_ * sizeof(bf16_t)); // 130 MB
  bf16_t* wih0_bf = (bf16_t*)carve((size_t)G_ * RIN_ * sizeof(bf16_t));      // 12 MB
  bf16_t* whh_bf[3];
  bf16_t* wih_bf[3];
  wih_bf[0] = wih0_bf;
  whh_bf[0] = (bf16_t*)carve((size_t)G_ * H_ * sizeof(bf16_t));
  wih_bf[1] = (bf16_t*)carve((size_t)G_ * H_ * sizeof(bf16_t));
  whh_bf[1] = (bf16_t*)carve((size_t)G_ * H_ * sizeof(bf16_t));
  wih_bf[2] = (bf16_t*)carve((size_t)G_ * H_ * sizeof(bf16_t));
  whh_bf[2] = (bf16_t*)carve((size_t)G_ * H_ * sizeof(bf16_t));
  float*  gi0  = (float*)carve((size_t)T_ * B_ * G_ * sizeof(float));        // 100 MB
  float*  stf  = (float*)carve((size_t)3 * 2 * BH_ * sizeof(float));
  bf16_t* stbf = (bf16_t*)carve((size_t)3 * 2 * BH_ * sizeof(bf16_t));
  unsigned* bar = (unsigned*)carve(256);

  // ---- weights -> bf16 ----
  {
    int n0 = G_ * RIN_;
    f32_to_bf16_k<<<(n0 + 255) / 256, 256, 0, stream>>>(w_ih[0], wih0_bf, n0);
    int n1 = G_ * H_;
    f32_to_bf16_k<<<(n1 + 255) / 256, 256, 0, stream>>>(w_hh[0], whh_bf[0], n1);
    f32_to_bf16_k<<<(n1 + 255) / 256, 256, 0, stream>>>(w_ih[1], wih_bf[1], n1);
    f32_to_bf16_k<<<(n1 + 255) / 256, 256, 0, stream>>>(w_hh[1], whh_bf[1], n1);
    f32_to_bf16_k<<<(n1 + 255) / 256, 256, 0, stream>>>(w_ih[2], wih_bf[2], n1);
    f32_to_bf16_k<<<(n1 + 255) / 256, 256, 0, stream>>>(w_hh[2], whh_bf[2], n1);
  }

  // ---- conv -> xs bf16 ----
  {
    int total = T_ * B_ * RIN_;
    conv_to_xs<<<(total + 255) / 256, 256, 0, stream>>>(x, conv_w, conv_b, xs_bf);
  }

  // ---- hoisted big GEMM ----
  {
    dim3 grid((T_ * B_) / 64, G_ / 128);
    gemm_gi0<<<grid, 256, 0, stream>>>(xs_bf, wih0_bf, b_ih[0], gi0);
  }

  // ---- init hidden state (slot 0) fp32 + bf16 mirror; reset barrier ----
  for (int l = 0; l < 3; ++l) {
    hipMemcpyAsync(stf + (size_t)(l * 2) * BH_, hinit[l], (size_t)BH_ * sizeof(float),
                   hipMemcpyDeviceToDevice, stream);
    f32_to_bf16_k<<<(BH_ + 255) / 256, 256, 0, stream>>>(hinit[l],
                                                         stbf + (size_t)(l * 2) * BH_, BH_);
  }
  hipMemsetAsync(bar, 0, 256, stream);

  // ---- persistent recurrence: 1 kernel, 256 steps, 240KB LDS weight cache ----
  gru_persistent<<<NWG_, 128, LDS_BYTES, stream>>>(
      whh_bf[0], wih_bf[1], whh_bf[1], wih_bf[2], whh_bf[2],
      b_hh[0], b_ih[1], b_hh[1], b_ih[2], b_hh[2],
      gi0, stf, stbf, out, bar);
}
